// PopulationGraphGNN_4346506903773
// MI455X (gfx1250) — matrix operand order repaired and checked by reading, not compiled, not run on hardware
//
#include <hip/hip_runtime.h>
#include <hip/hip_bf16.h>

#define N_NODES 100000
#define N_EDGES 1600000
#define XCOLS   1010
#define SUBJ    1000
#define DEMO    10
#define KPAD1   1024  // SUBJ padded to multiple of 32
#define COMBK   288   // 266 padded to multiple of 32

typedef __attribute__((ext_vector_type(16))) __bf16 v16bf;
typedef __attribute__((ext_vector_type(8)))  __bf16 v8bf;
typedef __attribute__((ext_vector_type(2)))  __bf16 v2bf;
typedef __attribute__((ext_vector_type(8)))  float  v8f;

// ---------------------------------------------------------------------------
// Tiled bf16 WMMA GEMM:  C = act(A @ Bt^T + bias)
//   A : M x K bf16 row-major (lda, K multiple of 32, lda multiple of 8)
//   Bt: N x K bf16 row-major (pre-transposed weights, ldb = K)
//   Block tile 128x128, 8 waves (4M x 2N), wave tile 32x64 = 2x4 frags,
//   k-step 32 via v_wmma_f32_16x16x32_bf16, f32 accumulation.
//   Double-buffered LDS (1 barrier/step) + register prefetch pipeline.
// ---------------------------------------------------------------------------
template<bool RELU, bool OUT_BF16>
__global__ __launch_bounds__(256)
void gemm_wmma_bf16(const __bf16* __restrict__ A, int lda,
                    const __bf16* __restrict__ Bt, int ldb, int K,
                    const float* __restrict__ bias,
                    void* __restrict__ Cout, int ldc, int M)
{
    constexpr int BM = 128, BN = 128;
    constexpr int LK = 40; // padded LDS row (80B: 16B aligned, conflict-free)
    __shared__ __bf16 As[2][BM * LK];
    __shared__ __bf16 Bs[2][BN * LK];

    const int tid  = threadIdx.x;
    const int lane = tid & 31;
    const int wave = tid >> 5;
    const int wm   = (wave & 3) * 32;   // wave M offset
    const int wn   = (wave >> 2) * 64;  // wave N offset
    const int m0   = blockIdx.x * BM;
    const int n0   = blockIdx.y * BN;

    v8f acc[2][4];
#pragma unroll
    for (int i = 0; i < 2; ++i)
#pragma unroll
        for (int j = 0; j < 4; ++j)
            acc[i][j] = v8f{0.f,0.f,0.f,0.f,0.f,0.f,0.f,0.f};

    // ---- staging: 16B chunks; 128x32 tile = 512 chunks = 2 per thread
    const int ca1 = tid + 256;
    const int ar0 = tid >> 2, ao0 = (tid & 3) * 8;
    const int ar1 = ca1 >> 2, ao1 = (ca1 & 3) * 8;
    int gr0 = m0 + ar0; if (gr0 >= M) gr0 = M - 1;   // clamp: OOB rows discarded later
    int gr1 = m0 + ar1; if (gr1 >= M) gr1 = M - 1;
    const long ga0 = (long)gr0 * lda + ao0;
    const long ga1 = (long)gr1 * lda + ao1;
    const long gb0 = (long)(n0 + ar0) * ldb + ao0;   // N multiple of 128: no clamp
    const long gb1 = (long)(n0 + ar1) * ldb + ao1;

    // ---- fragment addressing per ISA §7.12.2 (wave32, 16-bit A 16x32, B 32x16)
    const int l15   = lane & 15;
    const int ahalf = lane >> 4;
    const int akb   = ahalf * 8;   // A K-base: lanes16-31 start at K=8
    const int bkb   = ahalf * 8;   // B K-base: lanes16-31 start at K=8 (same order as A)

    // preload tile 0, stage into buffer 0
    v8bf ra0 = *(const v8bf*)(A  + ga0);
    v8bf ra1 = *(const v8bf*)(A  + ga1);
    v8bf rb0 = *(const v8bf*)(Bt + gb0);
    v8bf rb1 = *(const v8bf*)(Bt + gb1);
    *(v8bf*)&As[0][ar0 * LK + ao0] = ra0;
    *(v8bf*)&As[0][ar1 * LK + ao1] = ra1;
    *(v8bf*)&Bs[0][ar0 * LK + ao0] = rb0;
    *(v8bf*)&Bs[0][ar1 * LK + ao1] = rb1;
    __syncthreads();

    const int nk = K >> 5;
    for (int t = 0; t < nk; ++t) {
        const int cur = t & 1;
        const bool more = (t + 1 < nk);
        if (more) {                    // global prefetch overlaps WMMAs below
            const int k0 = (t + 1) * 32;
            ra0 = *(const v8bf*)(A  + ga0 + k0);
            ra1 = *(const v8bf*)(A  + ga1 + k0);
            rb0 = *(const v8bf*)(Bt + gb0 + k0);
            rb1 = *(const v8bf*)(Bt + gb1 + k0);
        }

        const __bf16* as = As[cur];
        const __bf16* bs = Bs[cur];
        v16bf bfr[4];
#pragma unroll
        for (int fn = 0; fn < 4; ++fn) {
            const __bf16* bp = &bs[(wn + fn * 16 + l15) * LK + bkb];
            union { v16bf v; v8bf h[2]; } bu;
            bu.h[0] = *(const v8bf*)bp;
            bu.h[1] = *(const v8bf*)(bp + 16);
            bfr[fn] = bu.v;
        }
#pragma unroll
        for (int fm = 0; fm < 2; ++fm) {
            const __bf16* ap = &as[(wm + fm * 16 + l15) * LK + akb];
            union { v16bf v; v8bf h[2]; } au;
            au.h[0] = *(const v8bf*)ap;        // K = akb .. akb+7
            au.h[1] = *(const v8bf*)(ap + 16); // K = akb+16 .. akb+23
#pragma unroll
            for (int fn = 0; fn < 4; ++fn)
                acc[fm][fn] = __builtin_amdgcn_wmma_f32_16x16x32_bf16(
                    false, au.v, false, bfr[fn], (short)0, acc[fm][fn], false, false);
        }

        if (more) {                    // stage next tile into the other buffer
            __bf16* asn = As[cur ^ 1];
            __bf16* bsn = Bs[cur ^ 1];
            *(v8bf*)&asn[ar0 * LK + ao0] = ra0;
            *(v8bf*)&asn[ar1 * LK + ao1] = ra1;
            *(v8bf*)&bsn[ar0 * LK + ao0] = rb0;
            *(v8bf*)&bsn[ar1 * LK + ao1] = rb1;
        }
        __syncthreads();
    }

    // ---- epilogue: C layout = VGPR r -> M = r + 8*(lane>=16), N = lane&15
#pragma unroll
    for (int fm = 0; fm < 2; ++fm) {
        const int rbase = m0 + wm + fm * 16 + ahalf * 8;
#pragma unroll
        for (int fn = 0; fn < 4; ++fn) {
            const int col = n0 + wn + fn * 16 + l15;
            const float bb = bias ? bias[col] : 0.f;
#pragma unroll
            for (int r = 0; r < 8; ++r) {
                const int row = rbase + r;
                if (row < M) {
                    float v = acc[fm][fn][r] + bb;
                    if (RELU) v = fmaxf(v, 0.f);
                    if (OUT_BF16) ((__bf16*)Cout)[(long)row * ldc + col] = (__bf16)v;
                    else          ((float*) Cout)[(long)row * ldc + col] = v;
                }
            }
        }
    }
}

// ------------------------- prep kernels ------------------------------------
// x[:, :1000] -> bf16, K padded to 1024 with zeros (2 elems/thread)
__global__ void k_xpad(const float* __restrict__ x, __bf16* __restrict__ xp) {
    int i = blockIdx.x * blockDim.x + threadIdx.x;   // over N_NODES*512 pairs
    if (i < N_NODES * 512) {
        const int node = i >> 9, k2 = (i & 511) * 2;
        v2bf o; o[0] = (__bf16)0.f; o[1] = (__bf16)0.f;
        if (k2 < SUBJ) {
            const float* p = &x[(long)node * XCOLS + k2];
            o[0] = (__bf16)p[0];
            o[1] = (__bf16)p[1];
        }
        *(v2bf*)&xp[(long)node * KPAD1 + k2] = o;
    }
}

// W (Kin x Nin fp32) -> out (Nin x Kpad bf16), transposed + K-padded
__global__ void k_transpose_pad(const float* __restrict__ W, __bf16* __restrict__ out,
                                int Kin, int Nin, int Kpad) {
    int i = blockIdx.x * blockDim.x + threadIdx.x;   // over Nin*Kpad
    if (i < Nin * Kpad) {
        const int n = i / Kpad, k = i - n * Kpad;
        out[i] = (k < Kin) ? (__bf16)W[(long)k * Nin + n] : (__bf16)0.f;
    }
}

// comb demo columns (256..265) + zero pad (266..287)
__global__ void k_fill_demo(const float* __restrict__ x, __bf16* __restrict__ comb) {
    int i = blockIdx.x * blockDim.x + threadIdx.x;   // N_NODES*32
    if (i < N_NODES * 32) {
        const int node = i >> 5, j = i & 31;
        __bf16 v = (__bf16)0.f;
        if (j < DEMO) v = (__bf16)x[(long)node * XCOLS + SUBJ + j];
        comb[(long)node * COMBK + 256 + j] = v;
    }
}

__global__ void k_fill_f32(float* __restrict__ p, float val, int n) {
    int i = blockIdx.x * blockDim.x + threadIdx.x;
    if (i < n) p[i] = val;
}

// ------------------------- graph kernels -----------------------------------
__global__ void k_deg(const long long* __restrict__ ei, float* __restrict__ deg) {
    int e = blockIdx.x * blockDim.x + threadIdx.x;
    if (e < N_EDGES) unsafeAtomicAdd(&deg[(int)ei[N_EDGES + e]], 1.0f);
}

__global__ void k_dinv(const float* __restrict__ deg, float* __restrict__ dinv) {
    int i = blockIdx.x * blockDim.x + threadIdx.x;
    if (i < N_NODES) dinv[i] = rsqrtf(deg[i]);
}

__global__ void k_agg_init(const float* __restrict__ xt, const float* __restrict__ dinv,
                           float* __restrict__ agg) {
    int i = blockIdx.x * blockDim.x + threadIdx.x;   // N_NODES*128
    if (i < N_NODES * 128) {
        const int node = i >> 7;
        const float di = dinv[node];
        agg[i] = xt[i] * di * di;   // self-loop contribution
    }
}

__global__ void k_scatter(const long long* __restrict__ ei, const float* __restrict__ xt,
                          const float* __restrict__ dinv, float* __restrict__ agg) {
    int t = blockIdx.x * blockDim.x + threadIdx.x;   // N_EDGES*128
    const int e = t >> 7, f = t & 127;
    if (e < N_EDGES) {
        const int s = (int)ei[e];
        const int d = (int)ei[N_EDGES + e];
        const float norm = dinv[s] * dinv[d];
        unsafeAtomicAdd(&agg[(long)d * 128 + f], xt[(long)s * 128 + f] * norm);
    }
}

__global__ __launch_bounds__(128)
void k_pool(const float* __restrict__ agg, const float* __restrict__ bg,
            float* __restrict__ pooled) {
    const int f = threadIdx.x;
    const int start = blockIdx.x * 256;
    int end = start + 256; if (end > N_NODES) end = N_NODES;
    const float b = bg[f];
    float s = 0.f;
    for (int i = start; i < end; ++i)
        s += fmaxf(agg[(long)i * 128 + f] + b, 0.f);
    unsafeAtomicAdd(&pooled[f], s);
}

__global__ __launch_bounds__(128)
void k_final(const float* __restrict__ pooled, const float* __restrict__ Wo,
             const float* __restrict__ bo, float* __restrict__ out) {
    __shared__ float red[128];
    const int f = threadIdx.x;
    red[f] = (pooled[f] / (float)N_NODES) * Wo[f];
    __syncthreads();
    for (int s = 64; s > 0; s >>= 1) {
        if (f < s) red[f] += red[f + s];
        __syncthreads();
    }
    if (f == 0) out[0] = red[0] + bo[0];
}

// ---------------------------------------------------------------------------
extern "C" void kernel_launch(void* const* d_in, const int* in_sizes, int n_in,
                              void* d_out, int out_size, void* d_ws, size_t ws_size,
                              hipStream_t stream) {
    const float*     x  = (const float*)d_in[0];
    const long long* ei = (const long long*)d_in[1];   // int64 edge_index (2, E)
    const float*     W1 = (const float*)d_in[2];
    const float*     b1 = (const float*)d_in[3];
    const float*     W2 = (const float*)d_in[4];
    const float*     b2 = (const float*)d_in[5];
    const float*     Wg = (const float*)d_in[6];
    const float*     bg = (const float*)d_in[7];
    const float*     Wo = (const float*)d_in[8];
    const float*     bo = (const float*)d_in[9];
    float* out = (float*)d_out;

    char* ws = (char*)d_ws;
    size_t off = 0;
    auto alloc = [&](size_t bytes) -> void* {
        void* p = ws + off;
        off += (bytes + 255) & ~(size_t)255;
        return p;
    };
    __bf16* w1t    = (__bf16*)alloc((size_t)512 * KPAD1 * 2);   // 512 x 1024
    __bf16* w2t    = (__bf16*)alloc((size_t)256 * 512 * 2);     // 256 x 512
    __bf16* wgt    = (__bf16*)alloc((size_t)128 * COMBK * 2);   // 128 x 288
    float*  deg    = (float*)alloc((size_t)N_NODES * 4);
    float*  dinv   = (float*)alloc((size_t)N_NODES * 4);
    float*  pooled = (float*)alloc(128 * 4);
    __bf16* hbuf   = (__bf16*)alloc((size_t)N_NODES * 512 * 2);
    __bf16* comb   = (__bf16*)alloc((size_t)N_NODES * COMBK * 2);
    // region shared by xpad (GEMM1 input) and xt+agg (dead/alive disjoint)
    char*   region = (char*)alloc((size_t)N_NODES * KPAD1 * 2); // 204.8 MB
    __bf16* xpad   = (__bf16*)region;
    float*  xt     = (float*)region;                            // 51.2 MB
    float*  agg    = (float*)(region + (size_t)N_NODES * 128 * 4);

    // ---- prep: pad/convert activations, transpose+pad weights
    k_xpad<<<(N_NODES * 512 + 255) / 256, 256, 0, stream>>>(x, xpad);
    k_transpose_pad<<<(512 * KPAD1 + 255) / 256, 256, 0, stream>>>(W1, w1t, SUBJ, 512, KPAD1);
    k_transpose_pad<<<(256 * 512 + 255) / 256, 256, 0, stream>>>(W2, w2t, 512, 256, 512);
    k_transpose_pad<<<(128 * COMBK + 255) / 256, 256, 0, stream>>>(Wg, wgt, 266, 128, COMBK);

    const int MB = (N_NODES + 127) / 128;
    // GEMM1: h = relu(x_subj @ W1 + b1) -> bf16
    gemm_wmma_bf16<true, true><<<dim3(MB, 512 / 128), 256, 0, stream>>>(
        xpad, KPAD1, w1t, KPAD1, KPAD1, b1, hbuf, 512, N_NODES);
    // comb demo columns
    k_fill_demo<<<(N_NODES * 32 + 255) / 256, 256, 0, stream>>>(x, comb);
    // GEMM2: subj = h @ W2 + b2 -> comb[:, :256] bf16
    gemm_wmma_bf16<false, true><<<dim3(MB, 256 / 128), 256, 0, stream>>>(
        hbuf, 512, w2t, 512, 512, b2, comb, COMBK, N_NODES);
    // GEMM3: xt = comb @ Wg_pad -> fp32 (overwrites dead xpad region)
    gemm_wmma_bf16<false, false><<<dim3(MB, 128 / 128), 256, 0, stream>>>(
        comb, COMBK, wgt, COMBK, COMBK, nullptr, xt, 128, N_NODES);

    // ---- graph aggregation
    k_fill_f32<<<(N_NODES + 255) / 256, 256, 0, stream>>>(deg, 1.0f, N_NODES); // self-loops
    k_deg<<<(N_EDGES + 255) / 256, 256, 0, stream>>>(ei, deg);
    k_dinv<<<(N_NODES + 255) / 256, 256, 0, stream>>>(deg, dinv);
    k_agg_init<<<(N_NODES * 128 + 255) / 256, 256, 0, stream>>>(xt, dinv, agg);
    {
        const long nt = (long)N_EDGES * 128;
        k_scatter<<<(unsigned)((nt + 255) / 256), 256, 0, stream>>>(ei, xt, dinv, agg);
    }
    // ---- pooling + output
    k_fill_f32<<<1, 128, 0, stream>>>(pooled, 0.f, 128);
    k_pool<<<(N_NODES + 255) / 256, 128, 0, stream>>>(agg, bg, pooled);
    k_final<<<1, 128, 0, stream>>>(pooled, Wo, bo, out);
}
